// GraphSAGE_86749749444804
// MI455X (gfx1250) — matrix operand order repaired and checked by reading, not compiled, not run on hardware
//
#include <hip/hip_runtime.h>
#include <stddef.h>
#include <stdint.h>

// ---------------------------------------------------------------------------
// GraphSAGE (2 layers) for MI455X / gfx1250.
//   agg  : edge gather + f32 global-atomic scatter-add (L2-resident)
//   gemm : [h | agg*deg_inv] @ [W_self ; W_neigh]  via v_wmma_f32_16x16x32_bf16
// A/B are pre-packed in bf16 into the exact CDNA5 WMMA fragment layouts so the
// inner loop is pure contiguous 32B loads + WMMA.
// ---------------------------------------------------------------------------

typedef __attribute__((ext_vector_type(16))) __bf16 v16bf;
typedef __attribute__((ext_vector_type(8)))  float  v8f;

#define H_FEATS 256
#define CHUNK_KB 4                      // 4 * 32 = 128 K-values staged in LDS
#define LDS_USHORTS (CHUNK_KB * 16 * 512)   // 32768 ushorts = 64 KB

__device__ __forceinline__ unsigned short f32_to_bf16(float f) {
    union { float f; uint32_t u; } v; v.f = f;
    uint32_t x = v.u;
    // round-to-nearest-even
    uint32_t r = x + 0x7FFFu + ((x >> 16) & 1u);
    return (unsigned short)(r >> 16);
}

// ---------------------------------------------------------------------------
__global__ __launch_bounds__(256) void fill_zero_kernel(float* __restrict__ p, long n) {
    long i = (long)blockIdx.x * blockDim.x + threadIdx.x;
    long stride = (long)gridDim.x * blockDim.x;
    for (; i < n; i += stride) p[i] = 0.0f;
}

__global__ __launch_bounds__(256) void degree_kernel(const int* __restrict__ dst,
                                                     float* __restrict__ deg, int E) {
    int e = blockIdx.x * blockDim.x + threadIdx.x;
    if (e < E) atomicAdd(&deg[dst[e]], 1.0f);
}

__global__ __launch_bounds__(256) void deg_inv_kernel(const float* __restrict__ deg,
                                                      float* __restrict__ dinv, int N) {
    int i = blockIdx.x * blockDim.x + threadIdx.x;
    if (i < N) dinv[i] = 1.0f / fmaxf(deg[i], 1.0f);
}

// One wave per edge; lane handles d/32 consecutive floats (float4 gathers,
// scalar f32 global atomics into the destination row).
__global__ __launch_bounds__(256) void aggregate_kernel(const float* __restrict__ h,
                                                        const int* __restrict__ src,
                                                        const int* __restrict__ dst,
                                                        float* __restrict__ agg,
                                                        int E, int d) {
    int gid  = blockIdx.x * blockDim.x + threadIdx.x;
    int edge = gid >> 5;
    int lane = gid & 31;
    if (edge >= E) return;
    int s = src[edge];
    int t = dst[edge];
    int per = d >> 5;                                  // 4 (d=128) or 8 (d=256)
    const float* hrow = h   + (size_t)s * d + lane * per;
    float*       arow = agg + (size_t)t * d + lane * per;
#pragma unroll 2
    for (int j = 0; j < per; j += 4) {
        float4 v = *reinterpret_cast<const float4*>(hrow + j);
        atomicAdd(arow + j + 0, v.x);
        atomicAdd(arow + j + 1, v.y);
        atomicAdd(arow + j + 2, v.z);
        atomicAdd(arow + j + 3, v.w);
    }
}

// Pack A = [h_self | agg*deg_inv]  (N x 2d f32) -> bf16, pre-swizzled into the
// CDNA5 16-bit A-fragment layout:  lane = 16*hi + (m%16),
//   element e in the lane's 16-value chunk maps to k = e + 8*((e>=8) + hi).
// Storage: Apack[mtile][kb][lane][e] contiguous -> lane reads 32B per fragment.
__global__ __launch_bounds__(256) void pack_A_kernel(const float* __restrict__ h_self,
                                                     const float* __restrict__ agg,
                                                     const float* __restrict__ dinv,
                                                     unsigned short* __restrict__ Apack,
                                                     int N, int d, int Kb, long total) {
    long t = (long)blockIdx.x * blockDim.x + threadIdx.x;
    if (t >= total) return;
    int e     = (int)(t & 15);
    int lane  = (int)((t >> 4) & 31);
    int kb    = (int)((t >> 9) % Kb);
    int mtile = (int)(t / (512L * Kb));
    int hi = lane >> 4;
    int m  = mtile * 16 + (lane & 15);
    int k  = kb * 32 + e + 8 * (((e >> 3) & 1) + hi);
    float val = 0.0f;
    if (m < N) {
        if (k < d) val = h_self[(size_t)m * d + k];
        else       val = agg[(size_t)m * d + (k - d)] * dinv[m];
    }
    Apack[t] = f32_to_bf16(val);
}

// Pack B = [W_self ; W_neigh]  (2d x 256 f32) -> bf16, B-fragment layout:
//   lane = 16*hi + n%16,  k = e + 8*((e>=8) + hi) within a 32-K block.
// Storage: Bpack[kb][ntile][lane][e] contiguous.
__global__ __launch_bounds__(256) void pack_B_kernel(const float* __restrict__ W_self,
                                                     const float* __restrict__ W_neigh,
                                                     unsigned short* __restrict__ Bpack,
                                                     int d, long total) {
    long t = (long)blockIdx.x * blockDim.x + threadIdx.x;
    if (t >= total) return;
    int e    = (int)(t & 15);
    int lane = (int)((t >> 4) & 31);
    int nt   = (int)((t >> 9) & 15);
    int kb   = (int)(t >> 13);
    int hi = lane >> 4;
    int k  = kb * 32 + e + 8 * (((e >> 3) & 1) + hi);
    int n  = nt * 16 + (lane & 15);
    float val = (k < d) ? W_self[(size_t)k * H_FEATS + n]
                        : W_neigh[(size_t)(k - d) * H_FEATS + n];
    Bpack[t] = f32_to_bf16(val);
}

// GEMM: out(N x 256) = A(N x K) @ B(K x 256) + bias, optional ReLU.
// 256 threads = 8 waves; wave w: M-tile = 4*blockIdx.x + (w>>1),
// columns nhalf = w&1 (128 cols = 8 accumulator tiles of 16x16).
// B staged 128-K-deep (64 KB) in LDS, shared by all waves.
__global__ __launch_bounds__(256) void gemm_wmma_kernel(const unsigned short* __restrict__ Apack,
                                                        const unsigned short* __restrict__ Bpack,
                                                        const float* __restrict__ bias,
                                                        float* __restrict__ out,
                                                        int Mtiles, int Kb_total, int N,
                                                        int relu) {
    extern __shared__ unsigned short smem[];          // LDS_USHORTS entries (64 KB)

    const int tid   = threadIdx.x;
    const int wave  = tid >> 5;
    const int lane  = tid & 31;
    const int mtile = blockIdx.x * 4 + (wave >> 1);
    const int nhalf = wave & 1;
    const bool valid = (mtile < Mtiles);

    v8f acc[8];
#pragma unroll
    for (int nt = 0; nt < 8; ++nt)
        acc[nt] = (v8f){0.f, 0.f, 0.f, 0.f, 0.f, 0.f, 0.f, 0.f};

    const int num_chunks = Kb_total / CHUNK_KB;
    for (int chunk = 0; chunk < num_chunks; ++chunk) {
        // ---- cooperative stage of 64 KB of pre-swizzled B into LDS ----
        {
            const uint4* gsrc = reinterpret_cast<const uint4*>(
                Bpack + (size_t)chunk * LDS_USHORTS);
            uint4* ldst = reinterpret_cast<uint4*>(smem);
#pragma unroll 4
            for (int i = tid; i < LDS_USHORTS / 8; i += 256)
                ldst[i] = gsrc[i];
        }
        __syncthreads();

        if (valid) {
#pragma unroll
            for (int kbl = 0; kbl < CHUNK_KB; ++kbl) {
                const int kb = chunk * CHUNK_KB + kbl;
                v16bf a = *reinterpret_cast<const v16bf*>(
                    Apack + (((size_t)mtile * Kb_total + kb) * 32 + lane) * 16);
#pragma unroll
                for (int nt = 0; nt < 8; ++nt) {
                    const int ntg = nhalf * 8 + nt;
                    v16bf b = *reinterpret_cast<const v16bf*>(
                        smem + ((size_t)(kbl * 16 + ntg) * 32 + lane) * 16);
                    acc[nt] = __builtin_amdgcn_wmma_f32_16x16x32_bf16(
                        false, a, false, b, (short)0, acc[nt], false, false);
                }
            }
        }
        __syncthreads();
    }

    if (!valid) return;

    // Epilogue: C layout -> lane holds (m = r + 8*hi, n = lane%16) per VGPR r.
    const int nl = lane & 15;
    const int hi = lane >> 4;
#pragma unroll
    for (int nt = 0; nt < 8; ++nt) {
        const int n  = nhalf * 128 + nt * 16 + nl;
        const float bv = bias[n];
#pragma unroll
        for (int r = 0; r < 8; ++r) {
            int m = mtile * 16 + 8 * hi + r;
            if (m < N) {
                float v = acc[nt][r] + bv;
                if (relu) v = fmaxf(v, 0.0f);
                out[(size_t)m * H_FEATS + n] = v;
            }
        }
    }
}

// ---------------------------------------------------------------------------
static inline size_t align256(size_t x) { return (x + 255) & ~(size_t)255; }

extern "C" void kernel_launch(void* const* d_in, const int* in_sizes, int n_in,
                              void* d_out, int out_size, void* d_ws, size_t ws_size,
                              hipStream_t stream) {
    const float* in_feat  = (const float*)d_in[0];
    const int*   src      = (const int*)d_in[1];
    const int*   dst      = (const int*)d_in[2];
    const float* W_self1  = (const float*)d_in[3];
    const float* W_neigh1 = (const float*)d_in[4];
    const float* b1       = (const float*)d_in[5];
    const float* W_self2  = (const float*)d_in[6];
    const float* W_neigh2 = (const float*)d_in[7];
    const float* b2       = (const float*)d_in[8];
    float* out = (float*)d_out;

    const int IN_F = 128;
    const int N = in_sizes[0] / IN_F;     // 50000
    const int E = in_sizes[1];            // 800000
    const int Mtiles = (N + 15) / 16;     // 3125

    // workspace layout
    char* ws = (char*)d_ws;
    size_t off = 0;
    float* deg  = (float*)(ws + off); off += align256((size_t)N * 4);
    float* dinv = (float*)(ws + off); off += align256((size_t)N * 4);
    float* h1   = (float*)(ws + off); off += align256((size_t)N * H_FEATS * 4);
    float* agg  = (float*)(ws + off); off += align256((size_t)N * H_FEATS * 4);   // reused L1/L2
    unsigned short* Apack = (unsigned short*)(ws + off);
    off += align256((size_t)Mtiles * 16 * 512 * 2);                               // up to K=512 bf16
    unsigned short* Bpack = (unsigned short*)(ws + off);
    off += align256((size_t)512 * H_FEATS * 2);
    (void)ws_size; (void)n_in; (void)out_size;

    const int TB = 256;
    dim3 blk(TB);
    auto blocks = [](long n, int tb) { return dim3((unsigned)((n + tb - 1) / tb)); };

    // ---- degrees ----
    fill_zero_kernel<<<blocks(N, TB), blk, 0, stream>>>(deg, N);
    degree_kernel<<<blocks(E, TB), blk, 0, stream>>>(dst, deg, E);
    deg_inv_kernel<<<blocks(N, TB), blk, 0, stream>>>(deg, dinv, N);

    // ================= Layer 1 (d = 128, K = 256) =================
    {
        const int d = 128, Kb = (2 * d) / 32;          // Kb = 8
        fill_zero_kernel<<<blocks((long)N * d, TB), blk, 0, stream>>>(agg, (long)N * d);
        aggregate_kernel<<<blocks((long)E * 32, TB), blk, 0, stream>>>(
            in_feat, src, dst, agg, E, d);

        long totA = (long)Mtiles * Kb * 512;
        pack_A_kernel<<<blocks(totA, TB), blk, 0, stream>>>(
            in_feat, agg, dinv, Apack, N, d, Kb, totA);
        long totB = (long)Kb * 16 * 512;
        pack_B_kernel<<<blocks(totB, TB), blk, 0, stream>>>(
            W_self1, W_neigh1, Bpack, d, totB);

        gemm_wmma_kernel<<<blocks((long)Mtiles, 4), blk, LDS_USHORTS * 2, stream>>>(
            Apack, Bpack, b1, h1, Mtiles, Kb, N, /*relu=*/1);
    }

    // ================= Layer 2 (d = 256, K = 512) =================
    {
        const int d = 256, Kb = (2 * d) / 32;          // Kb = 16
        fill_zero_kernel<<<blocks((long)N * d, TB), blk, 0, stream>>>(agg, (long)N * d);
        aggregate_kernel<<<blocks((long)E * 32, TB), blk, 0, stream>>>(
            h1, src, dst, agg, E, d);

        long totA = (long)Mtiles * Kb * 512;
        pack_A_kernel<<<blocks(totA, TB), blk, 0, stream>>>(
            h1, agg, dinv, Apack, N, d, Kb, totA);
        long totB = (long)Kb * 16 * 512;
        pack_B_kernel<<<blocks(totB, TB), blk, 0, stream>>>(
            W_self2, W_neigh2, Bpack, d, totB);

        gemm_wmma_kernel<<<blocks((long)Mtiles, 4), blk, LDS_USHORTS * 2, stream>>>(
            Apack, Bpack, b2, out, Mtiles, Kb, N, /*relu=*/0);
    }
}
